// CNN2_P_3719441679179
// MI455X (gfx1250) — matrix operand order repaired and checked by reading, not compiled, not run on hardware
//
#include <hip/hip_runtime.h>
#include <hip/hip_bf16.h>

// ---------------------------------------------------------------------------
// CDNA5 (gfx1250, wave32) bf16 WMMA pipeline for the 5-stage CNN.
// Every stage is a GEMM on v_wmma_f32_16x16x32_bf16 with fused bias (+ReLU except fc2).
// MODE 1/2 GEMMs use GLOBAL_LOAD_ASYNC_TO_LDS double-buffering when the
// toolchain exposes the gfx1250 async builtins (guarded by __has_builtin).
// ---------------------------------------------------------------------------

typedef __attribute__((ext_vector_type(4)))  float   v4f;
typedef __attribute__((ext_vector_type(8)))  float   v8f;
typedef __attribute__((ext_vector_type(4)))  int     v4i;
typedef __attribute__((ext_vector_type(4)))  __bf16  v4bf;
typedef __attribute__((ext_vector_type(8)))  __bf16  v8bf;
typedef __attribute__((ext_vector_type(16))) __bf16  v16bf;

#if __has_builtin(__builtin_amdgcn_global_load_async_to_lds_b128) && \
    __has_builtin(__builtin_amdgcn_s_wait_asynccnt)
#define HAVE_ASYNC_LDS 1
#else
#define HAVE_ASYNC_LDS 0
#endif

__device__ __forceinline__ __bf16 f2bf(float f) {
  unsigned u = __builtin_bit_cast(unsigned, f);
  u += 0x7FFFu + ((u >> 16) & 1u);                 // round-to-nearest-even
  unsigned short h = (unsigned short)(u >> 16);
  return __builtin_bit_cast(__bf16, h);
}

__device__ __forceinline__ void store_val(__bf16* p, float v) { *p = f2bf(v); }
__device__ __forceinline__ void store_val(float*  p, float v) { *p = v; }

#define CAT8TO16(lo, hi) \
  __builtin_shufflevector((lo), (hi), 0,1,2,3,4,5,6,7,8,9,10,11,12,13,14,15)

#if HAVE_ASYNC_LDS
// 16B global -> LDS, tracked by ASYNCcnt (GLOBAL_LOAD_ASYNC_TO_LDS_B128).
// Param 1 is an AS1 int4* per the compiler diagnostic; param 2 is the LDS
// destination in AS3.
__device__ __forceinline__ void async_cp16(const __bf16* g, __bf16* l) {
  __builtin_amdgcn_global_load_async_to_lds_b128(
      (__attribute__((address_space(1))) v4i*)(const_cast<__bf16*>(g)),
      (__attribute__((address_space(3))) v4i*)(l),
      0, 0);
}
#endif

// ---------------------------------------------------------------------------
// Weight re-layout kernels (run once per launch; tiny vs. GEMM cost).
// All produce bf16 "Bt" matrices in N-major (N x K) layout so the GEMM stages
// them into LDS with contiguous 16B chunks (no transpose in the hot loop).
// ---------------------------------------------------------------------------

// w1 (256,64,2) -> w1r[c][k], k<64: tap1 (xi), k>=64: tap0 (x0).  K=128.
__global__ void prep_w1_k(const float* __restrict__ w1, __bf16* __restrict__ out) {
  int idx = blockIdx.x * blockDim.x + threadIdx.x;
  if (idx >= 256 * 128) return;
  int c = idx >> 7, k = idx & 127;
  float v = (k < 64) ? w1[(c * 64 + k) * 2 + 1]
                     : w1[(c * 64 + (k - 64)) * 2 + 0];
  out[idx] = f2bf(v);
}

// w (256,256,3) -> wr[c][kk], kk = k*256 + c'  (matches (B,L,256) act layout)
__global__ void prep_convw_k(const float* __restrict__ w, __bf16* __restrict__ out) {
  int idx = blockIdx.x * blockDim.x + threadIdx.x;
  if (idx >= 256 * 768) return;
  int c = idx / 768, kk = idx % 768;
  int k = kk >> 8, cp = kk & 255;
  out[idx] = f2bf(w[(c * 256 + cp) * 3 + k]);
}

// wfc1 (1024,31488 c-major) -> wr[f][kk], kk = l*256 + c (l-major like y3)
__global__ void prep_wfc1_k(const float* __restrict__ w, __bf16* __restrict__ out) {
  int idx = blockIdx.x * blockDim.x + threadIdx.x;
  if (idx >= 1024 * 31488) return;
  int f = idx / 31488, kk = idx % 31488;
  int l = kk >> 8, c = kk & 255;
  out[idx] = f2bf(w[(size_t)f * 31488 + c * 123 + l]);
}

// copy-convert with zero padding (pads fc2's Bt to a full 128-row N tile so
// the GEMM B staging needs no bounds guard -> constant async wait count)
__global__ void prep_copy_pad_k(const float* __restrict__ w, __bf16* __restrict__ out,
                                int nsrc, int ntot) {
  int idx = blockIdx.x * blockDim.x + threadIdx.x;
  if (idx >= ntot) return;
  out[idx] = (idx < nsrc) ? f2bf(w[idx]) : f2bf(0.f);
}

// ---------------------------------------------------------------------------
// Unified tiled GEMM:  C[m,n] = act( sum_k A[m,k]*Bt[n,k] + bias[n] )
//   MODE 0: A rows built on the fly from fp32 x (layer 1, K=128 augmented)
//   MODE 1: implicit conv GEMM, A row (b,l) -> Yprev + (b*Lin + l)*256, K=768
//   MODE 2: plain bf16 A, row stride K
// Block tile 128x128, 256 threads = 8 waves, each wave 32x64 (2x4 WMMA).
// Bt must have ceil(N/128)*128 rows (zero-padded); epilogue guards on N.
// ---------------------------------------------------------------------------
template <int MODE, bool RELU, typename OutT>
__global__ __launch_bounds__(256) void gemm_bf16_wmma(
    const float*  __restrict__ Xf,    // MODE 0 only
    const __bf16* __restrict__ Abf,   // MODE 1/2
    const __bf16* __restrict__ Bt,    // N x K, bf16 (N padded to 128 multiple)
    const float*  __restrict__ bias,  // length N
    OutT*         __restrict__ Cout,  // M x N
    int M, int N, int K, int Lout, int Lin)
{
  constexpr int LDSS = 40;            // 32 + 8 pad halves: 16B aligned, bank-spread
  __shared__ __bf16 As[2][128 * LDSS];
  __shared__ __bf16 Bs[2][128 * LDSS];

  const int tid  = threadIdx.x;
  const int lane = tid & 31;
  const int wave = tid >> 5;
  const int m0 = blockIdx.x * 128;
  const int n0 = blockIdx.y * 128;
  const int nK = K >> 5;

  // ---- staging assignments: 512 16B chunks per tile, 2 per thread ----
  const int sRow = tid >> 2;          // 0..63 (and +64)
  const int sKc  = (tid & 3) * 8;     // half offset within 32-half K slice

  const __bf16* bPtr0 = Bt + (size_t)(n0 + sRow) * K + sKc;
  const __bf16* bPtr1 = Bt + (size_t)(n0 + sRow + 64) * K + sKc;

  const __bf16* aPtr0 = nullptr; const __bf16* aPtr1 = nullptr;
  size_t xXi[4] = {0,0,0,0}; size_t xX0[4] = {0,0,0,0};
  int    xRow[4] = {0,0,0,0}; int xKcf = 0;

  if constexpr (MODE == 1) {          // implicit conv GEMM
    int gm = m0 + sRow;
    unsigned b = (unsigned)gm / (unsigned)Lout; int l = gm - (int)b * Lout;
    aPtr0 = Abf + ((size_t)b * Lin + l) * 256 + sKc;
    gm = m0 + sRow + 64;
    b = (unsigned)gm / (unsigned)Lout; l = gm - (int)b * Lout;
    aPtr1 = Abf + ((size_t)b * Lin + l) * 256 + sKc;
  } else if constexpr (MODE == 2) {
    aPtr0 = Abf + (size_t)(m0 + sRow) * K + sKc;
    aPtr1 = Abf + (size_t)(m0 + sRow + 64) * K + sKc;
  } else {                            // MODE 0: fp32 x, 4 float4 chunks/thread
    xKcf = (tid & 7) * 4;
    int rr = tid >> 3;                // 0..31
    for (int i = 0; i < 4; ++i) {
      int row = rr + 32 * i;
      int gm = m0 + row;
      unsigned b = (unsigned)gm / 127u; int l = gm - (int)b * 127;
      xRow[i] = row;
      xXi[i] = (size_t)b * 8192 + (size_t)(l + 1) * 64 + xKcf;  // xi tap
      xX0[i] = (size_t)b * 8192 + xKcf;                          // x0 tap
    }
  }

  const v8f vz = {0.f,0.f,0.f,0.f,0.f,0.f,0.f,0.f};
  v8f acc[2][4];
  for (int i = 0; i < 2; ++i) for (int j = 0; j < 4; ++j) acc[i][j] = vz;

  const int mw = (wave & 3) * 32;     // 4 waves along M
  const int nw = (wave >> 2) * 64;    // 2 waves along N
  const int l15  = lane & 15;
  const int hsel = lane >> 4;

  // ---- fragment load + 8x WMMA for one K slice (ISA 7.12.2 layouts) ----
  auto compute_step = [&](const __bf16* Ab, const __bf16* Bb) {
    // A 16x32: M = lane&15; K = 8*(lane>=16) + {0..7, 16..23}
    v16bf af[2];
    for (int i = 0; i < 2; ++i) {
      int r = mw + i * 16 + l15;
      v8bf lo = *(const v8bf*)&Ab[r * LDSS + hsel * 8];
      v8bf hi = *(const v8bf*)&Ab[r * LDSS + hsel * 8 + 16];
      af[i] = CAT8TO16(lo, hi);
    }
    // B 32x16: N = lane&15; K = 8*(lane>=16) + {0..7, 16..23}
    v16bf bfr[4];
    for (int j = 0; j < 4; ++j) {
      int r = nw + j * 16 + l15;
      v8bf lo = *(const v8bf*)&Bb[r * LDSS + hsel * 8];
      v8bf hi = *(const v8bf*)&Bb[r * LDSS + hsel * 8 + 16];
      bfr[j] = CAT8TO16(lo, hi);
    }
    for (int i = 0; i < 2; ++i)
      for (int j = 0; j < 4; ++j)
        acc[i][j] = __builtin_amdgcn_wmma_f32_16x16x32_bf16(
            false, af[i], false, bfr[j], (short)0, acc[i][j], false, false);
  };

#if HAVE_ASYNC_LDS
  if constexpr (MODE != 0) {
    // -------- async double-buffered pipeline (ASYNCcnt tracked) --------
    // Each thread issues exactly 4 x B128 per tile; async loads complete
    // in order, so waiting down to 4 outstanding (= next tile's loads)
    // guarantees the current tile has fully landed in LDS.
    auto issue_tile = [&](int ks) {
      int p = ks & 1;
      async_cp16(aPtr0 + (size_t)ks * 32, &As[p][sRow * LDSS + sKc]);
      async_cp16(aPtr1 + (size_t)ks * 32, &As[p][(sRow + 64) * LDSS + sKc]);
      async_cp16(bPtr0 + (size_t)ks * 32, &Bs[p][sRow * LDSS + sKc]);
      async_cp16(bPtr1 + (size_t)ks * 32, &Bs[p][(sRow + 64) * LDSS + sKc]);
    };
    issue_tile(0);
    for (int ks = 0; ks < nK; ++ks) {
      if (ks + 1 < nK) {
        issue_tile(ks + 1);                       // prefetch next tile
        __builtin_amdgcn_s_wait_asynccnt(4);      // current tile landed
      } else {
        __builtin_amdgcn_s_wait_asynccnt(0);
      }
      __syncthreads();                            // all waves' tiles landed
      compute_step(As[ks & 1], Bs[ks & 1]);
      __syncthreads();                            // buffer free for reuse
    }
  } else
#endif
  {
    // -------- synchronous single-buffer path (MODE 0 / fallback) --------
    for (int ks = 0; ks < nK; ++ks) {
      v8bf b0v = *(const v8bf*)(bPtr0 + (size_t)ks * 32);
      v8bf b1v = *(const v8bf*)(bPtr1 + (size_t)ks * 32);
      *(v8bf*)&Bs[0][sRow * LDSS + sKc]        = b0v;
      *(v8bf*)&Bs[0][(sRow + 64) * LDSS + sKc] = b1v;

      if constexpr (MODE == 0) {
        for (int i = 0; i < 4; ++i) {
          size_t off = (ks < 2) ? (xXi[i] + (size_t)ks * 32)
                                : (xX0[i] + (size_t)(ks - 2) * 32);
          v4f f = *(const v4f*)(Xf + off);
          v4bf h; h[0]=f2bf(f[0]); h[1]=f2bf(f[1]); h[2]=f2bf(f[2]); h[3]=f2bf(f[3]);
          *(v4bf*)&As[0][xRow[i] * LDSS + xKcf] = h;
        }
      } else {
        v8bf a0v = *(const v8bf*)(aPtr0 + (size_t)ks * 32);
        v8bf a1v = *(const v8bf*)(aPtr1 + (size_t)ks * 32);
        *(v8bf*)&As[0][sRow * LDSS + sKc]        = a0v;
        *(v8bf*)&As[0][(sRow + 64) * LDSS + sKc] = a1v;
        if (ks + 1 < nK) __builtin_prefetch(aPtr0 + (size_t)(ks + 1) * 32, 0, 1);
      }
      if (ks + 1 < nK) __builtin_prefetch(bPtr0 + (size_t)(ks + 1) * 32, 0, 1);

      __syncthreads();
      compute_step(As[0], Bs[0]);
      __syncthreads();
    }
  }

  // ---- epilogue: C layout M = r + 8*(lane>=16), N = lane&15 ----
  for (int j = 0; j < 4; ++j) {
    int gn = n0 + nw + j * 16 + l15;
    if (gn >= N) continue;
    float bv = bias[gn];
    for (int i = 0; i < 2; ++i) {
      for (int r = 0; r < 8; ++r) {
        int gm = m0 + mw + i * 16 + r + hsel * 8;
        float v = acc[i][j][r] + bv;
        if (RELU) v = v > 0.f ? v : 0.f;
        store_val(&Cout[(size_t)gm * N + gn], v);
      }
    }
  }
}

// ---------------------------------------------------------------------------
extern "C" void kernel_launch(void* const* d_in, const int* in_sizes, int n_in,
                              void* d_out, int out_size, void* d_ws, size_t ws_size,
                              hipStream_t stream) {
  (void)in_sizes; (void)n_in; (void)out_size; (void)ws_size;
  const float* x    = (const float*)d_in[0];
  const float* w1   = (const float*)d_in[1];
  const float* b1   = (const float*)d_in[2];
  const float* w2   = (const float*)d_in[3];
  const float* b2   = (const float*)d_in[4];
  const float* w3   = (const float*)d_in[5];
  const float* b3   = (const float*)d_in[6];
  const float* wfc1 = (const float*)d_in[7];
  const float* bfc1 = (const float*)d_in[8];
  const float* wfc2 = (const float*)d_in[9];
  const float* bfc2 = (const float*)d_in[10];

  char* ws = (char*)d_ws;
  size_t off = 0;
  auto alloc = [&](size_t bytes) -> void* {
    void* p = ws + off; off += (bytes + 255) & ~(size_t)255; return p;
  };
  __bf16* w1r  = (__bf16*)alloc((size_t)256 * 128 * 2);
  __bf16* w2r  = (__bf16*)alloc((size_t)256 * 768 * 2);
  __bf16* w3r  = (__bf16*)alloc((size_t)256 * 768 * 2);
  __bf16* wf1r = (__bf16*)alloc((size_t)1024 * 31488 * 2);
  __bf16* wf2r = (__bf16*)alloc((size_t)128 * 1024 * 2);   // padded N: 16 -> 128
  __bf16* yA   = (__bf16*)alloc((size_t)2048 * 127 * 256 * 2); // y1, reused for y3
  __bf16* yB   = (__bf16*)alloc((size_t)2048 * 125 * 256 * 2); // y2, reused for y4

  // weight re-layout (cheap)
  prep_w1_k      <<<(256 * 128 + 255) / 256, 256, 0, stream>>>(w1, w1r);
  prep_convw_k   <<<(256 * 768 + 255) / 256, 256, 0, stream>>>(w2, w2r);
  prep_convw_k   <<<(256 * 768 + 255) / 256, 256, 0, stream>>>(w3, w3r);
  prep_wfc1_k    <<<(1024 * 31488 + 255) / 256, 256, 0, stream>>>(wfc1, wf1r);
  prep_copy_pad_k<<<(128 * 1024 + 255) / 256, 256, 0, stream>>>(
      wfc2, wf2r, 16 * 1024, 128 * 1024);

  // layer 1: M=2048*127, N=256, K=128 (augmented with x0 tap) -> y1 (B,127,256)
  gemm_bf16_wmma<0, true, __bf16><<<dim3(260096 / 128, 2), 256, 0, stream>>>(
      x, nullptr, w1r, b1, yA, 260096, 256, 128, 127, 0);

  // conv2: implicit GEMM, M=2048*125, K=768 -> y2 (B,125,256)
  gemm_bf16_wmma<1, true, __bf16><<<dim3(256000 / 128, 2), 256, 0, stream>>>(
      nullptr, yA, w2r, b2, yB, 256000, 256, 768, 125, 127);

  // conv3: M=2048*123, K=768 -> y3 (B,123,256)  (reuses y1's buffer)
  gemm_bf16_wmma<1, true, __bf16><<<dim3(251904 / 128, 2), 256, 0, stream>>>(
      nullptr, yB, w3r, b3, yA, 251904, 256, 768, 123, 125);

  // fc1: M=2048, N=1024, K=31488 -> y4 (reuses y2's buffer)
  gemm_bf16_wmma<2, true, __bf16><<<dim3(2048 / 128, 8), 256, 0, stream>>>(
      nullptr, yA, wf1r, bfc1, yB, 2048, 1024, 31488, 0, 0);

  // fc2: M=2048, N=16 (Bt zero-padded to 128 rows), K=1024 -> d_out fp32
  gemm_bf16_wmma<2, false, float><<<dim3(2048 / 128, 1), 256, 0, stream>>>(
      nullptr, yB, wf2r, bfc2, (float*)d_out, 2048, 16, 1024, 0, 0);
}
